// MoELayer_90228672954431
// MI455X (gfx1250) — matrix operand order repaired and checked by reading, not compiled, not run on hardware
//
#include <hip/hip_runtime.h>
#include <hip/hip_bf16.h>
#include <math.h>

// Problem constants (match reference)
constexpr int NTOK = 8192;
constexpr int DD   = 2048;
constexpr int NE   = 8;

// GEMM tiling
constexpr int MT = 128;  // token rows per block
constexpr int NT = 64;   // output cols per block
constexpr int KT = 32;   // K per step (matches wmma 16x16x32)
constexpr int A_STRIDE = 40;  // bf16 elems per A row in LDS (80B, 16B-aligned)
constexpr int B_STRIDE = 72;  // bf16 elems per B row in LDS (144B, 16B-aligned)

typedef __attribute__((ext_vector_type(16))) __bf16 v16bf;
typedef __attribute__((ext_vector_type(8)))  __bf16 v8bf;
typedef __attribute__((ext_vector_type(4)))  __bf16 v4bf;
typedef __attribute__((ext_vector_type(8)))  float  v8f;
typedef __attribute__((ext_vector_type(4)))  float  v4f;

union BF16x16 { v16bf v; v8bf h[2]; };

// ---------------------------------------------------------------------------
// Kernel 0: reset per-expert token counters
// ---------------------------------------------------------------------------
__global__ void zero_cnt(int* cnt) {
    if (threadIdx.x < NE) cnt[threadIdx.x] = 0;
}

// ---------------------------------------------------------------------------
// Kernel P1: convert x (f32) -> xb (bf16), 8 elems/thread, b128 in / b128 out
// ---------------------------------------------------------------------------
__global__ __launch_bounds__(256) void cvt_x(const float* __restrict__ x,
                                             __bf16* __restrict__ xb)
{
    const size_t g = (size_t)blockIdx.x * 256 + threadIdx.x;
    const size_t base = g * 8;
    v4f f0 = *(const v4f*)(x + base);
    v4f f1 = *(const v4f*)(x + base + 4);
    v8bf r;
    r[0] = (__bf16)f0[0]; r[1] = (__bf16)f0[1];
    r[2] = (__bf16)f0[2]; r[3] = (__bf16)f0[3];
    r[4] = (__bf16)f1[0]; r[5] = (__bf16)f1[1];
    r[6] = (__bf16)f1[2]; r[7] = (__bf16)f1[3];
    *(v8bf*)(xb + base) = r;
}

// ---------------------------------------------------------------------------
// Kernel P2: convert + transpose W[e][o][i] (f32) -> Wt[e][i][o] (bf16)
// 64x64 tiles through LDS. grid = (D/64, D/64, E)
// ---------------------------------------------------------------------------
__global__ __launch_bounds__(256) void cvt_w(const float* __restrict__ W,
                                             __bf16* __restrict__ Wt)
{
    const int e  = blockIdx.z;
    const int i0 = blockIdx.x * 64;
    const int o0 = blockIdx.y * 64;
    const int t  = threadIdx.x;

    __shared__ __bf16 ltile[64 * 72];   // [i_local][o_local], padded

    // read phase: coalesced rows of W (contiguous in i)
    const int orow = t >> 4;            // 0..15
    const int icol = (t & 15) * 4;      // 0..60 step 4
    #pragma unroll
    for (int j = 0; j < 4; ++j) {
        const int o = orow + 16 * j;
        v4f f = *(const v4f*)(W + ((size_t)e * DD + (o0 + o)) * DD + i0 + icol);
        #pragma unroll
        for (int jj = 0; jj < 4; ++jj)
            ltile[(icol + jj) * 72 + o] = (__bf16)f[jj];
    }
    __syncthreads();

    // write phase: rows of Wt (contiguous in o)
    const int irow = t >> 4;
    const int ocol = (t & 15) * 4;
    #pragma unroll
    for (int j = 0; j < 4; ++j) {
        const int i = irow + 16 * j;
        v4bf r;
        r[0] = ltile[i * 72 + ocol];
        r[1] = ltile[i * 72 + ocol + 1];
        r[2] = ltile[i * 72 + ocol + 2];
        r[3] = ltile[i * 72 + ocol + 3];
        *(v4bf*)(Wt + ((size_t)e * DD + (i0 + i)) * DD + o0 + ocol) = r;
    }
}

// ---------------------------------------------------------------------------
// Kernel 1: router. One block per token, 8 wave32s, one expert logit per wave.
// Appends token to per-expert lists and initializes out[n,:] with the
// combine-weighted biases (out fully written each call).
// ---------------------------------------------------------------------------
__global__ __launch_bounds__(256) void router_kernel(
    const float* __restrict__ x, const float* __restrict__ Wr,
    const float* __restrict__ br, const float* __restrict__ b,
    float* __restrict__ out, int* __restrict__ cnt,
    int* __restrict__ tok, float* __restrict__ wgt)
{
    const int n    = blockIdx.x;
    const int tid  = threadIdx.x;
    const int wave = tid >> 5;
    const int lane = tid & 31;

    __shared__ float s_logit[NE];
    __shared__ int   s_e0, s_e1;
    __shared__ float s_w0, s_w1;

    const float* xr = x  + (size_t)n    * DD;
    const float* wr = Wr + (size_t)wave * DD;

    float part = 0.f;
    for (int j = lane; j < DD; j += 32)
        part += xr[j] * wr[j];
    #pragma unroll
    for (int off = 16; off > 0; off >>= 1)
        part += __shfl_xor(part, off, 32);          // wave32 reduction
    if (lane == 0) s_logit[wave] = part + br[wave];
    __syncthreads();

    if (tid == 0) {
        float p[NE];
        float mx = -INFINITY;
        #pragma unroll
        for (int e = 0; e < NE; ++e) { p[e] = s_logit[e]; mx = fmaxf(mx, p[e]); }
        float sum = 0.f;
        #pragma unroll
        for (int e = 0; e < NE; ++e) { p[e] = __expf(p[e] - mx); sum += p[e]; }
        const float inv = 1.f / sum;
        #pragma unroll
        for (int e = 0; e < NE; ++e) p[e] *= inv;

        int e0 = 0; float v0 = p[0];
        #pragma unroll
        for (int e = 1; e < NE; ++e) if (p[e] > v0) { v0 = p[e]; e0 = e; }
        int e1 = (e0 == 0) ? 1 : 0; float v1 = p[e1];
        #pragma unroll
        for (int e = 0; e < NE; ++e) if (e != e0 && p[e] > v1) { v1 = p[e]; e1 = e; }

        int pos0 = atomicAdd(&cnt[e0], 1);
        tok[e0 * NTOK + pos0] = n;  wgt[e0 * NTOK + pos0] = v0;
        int pos1 = atomicAdd(&cnt[e1], 1);
        tok[e1 * NTOK + pos1] = n;  wgt[e1 * NTOK + pos1] = v1;

        s_e0 = e0; s_e1 = e1; s_w0 = v0; s_w1 = v1;
    }
    __syncthreads();

    const float w0 = s_w0, w1 = s_w1;
    const float* b0p = b + (size_t)s_e0 * DD;
    const float* b1p = b + (size_t)s_e1 * DD;
    float* op = out + (size_t)n * DD;
    for (int d = tid; d < DD; d += 256)
        op[d] = w0 * b0p[d] + w1 * b1p[d];
}

// ---------------------------------------------------------------------------
// Kernel 2: grouped expert GEMM (bf16 WMMA, f32 accumulate).
// Block tile 128x64, 8 waves of 32x32 (4 WMMAs per wave per K-step),
// software-pipelined global->LDS staging, weighted atomic scatter-add.
// ---------------------------------------------------------------------------
__global__ __launch_bounds__(256) void moe_gemm(
    const __bf16* __restrict__ xb, const __bf16* __restrict__ Wt,
    const int* __restrict__ cnt, const int* __restrict__ tok,
    const float* __restrict__ wgt, float* __restrict__ out)
{
    const int e = blockIdx.z;
    const int c = cnt[e];
    const int m0 = blockIdx.x * MT;
    if (m0 >= c) return;
    const int o0 = blockIdx.y * NT;

    const int tid  = threadIdx.x;
    const int wave = tid >> 5;
    const int lane = tid & 31;
    const int wm   = wave & 3;    // 4 M sub-tiles of 32
    const int wn   = wave >> 2;   // 2 N sub-tiles of 32

    __shared__ __bf16 la[MT * A_STRIDE];   // A tile [m][k]
    __shared__ __bf16 lb[KT * B_STRIDE];   // B tile [k][n]
    __shared__ int    s_tok[MT];
    __shared__ float  s_wgt[MT];

    if (tid < MT) {
        const int gi = m0 + tid;
        if (gi < c) { s_tok[tid] = tok[e * NTOK + gi]; s_wgt[tid] = wgt[e * NTOK + gi]; }
        else        { s_tok[tid] = tok[e * NTOK];      s_wgt[tid] = 0.f; }
    }
    __syncthreads();

    // A staging: thread -> (row, 16-elem half);  B staging: thread -> (k, 8-elem group)
    const int ar = tid >> 1;                 // 0..127
    const int ah = (tid & 1) * 16;           // 0 or 16
    const int bk = tid >> 3;                 // 0..31
    const int bc = (tid & 7) * 8;            // 0..56 step 8

    const __bf16* agp = xb + (size_t)s_tok[ar] * DD + ah;
    const __bf16* bgp = Wt + ((size_t)e * DD + bk) * DD + o0 + bc;

    v8bf a_st0, a_st1, b_st;
    a_st0 = *(const v8bf*)(agp);
    a_st1 = *(const v8bf*)(agp + 8);
    b_st  = *(const v8bf*)(bgp);

    v8f acc00 = {}, acc01 = {}, acc10 = {}, acc11 = {};

    for (int k0 = 0; k0 < DD; k0 += KT) {
        // commit staged regs to LDS
        *(v8bf*)(&la[ar * A_STRIDE + ah])     = a_st0;
        *(v8bf*)(&la[ar * A_STRIDE + ah + 8]) = a_st1;
        *(v8bf*)(&lb[bk * B_STRIDE + bc])     = b_st;
        __syncthreads();

        // issue next step's global loads early (latency hidden by WMMAs)
        if (k0 + KT < DD) {
            const __bf16* ag = agp + k0 + KT;
            const __bf16* bg = bgp + (size_t)(k0 + KT) * DD;
            a_st0 = *(const v8bf*)(ag);
            a_st1 = *(const v8bf*)(ag + 8);
            b_st  = *(const v8bf*)(bg);
            __builtin_prefetch(bg + (size_t)KT * DD, 0, 1);  // global_prefetch_b8
        }

        // fragments per CDNA5 wave32 layouts
        const int ko = (lane < 16) ? 0 : 8;
        BF16x16 a0, a1, bf0, bf1;
        const __bf16* ar0 = &la[(wm * 32 + (lane & 15)) * A_STRIDE];
        const __bf16* ar1 = ar0 + 16 * A_STRIDE;
        a0.h[0] = *(const v8bf*)(ar0 + ko);
        a0.h[1] = *(const v8bf*)(ar0 + 16 + ko);
        a1.h[0] = *(const v8bf*)(ar1 + ko);
        a1.h[1] = *(const v8bf*)(ar1 + 16 + ko);
        const int bn = wn * 32 + (lane & 15);
        #pragma unroll
        for (int s = 0; s < 8; ++s) {
            bf0.v[s]     = lb[(ko + s) * B_STRIDE + bn];
            bf0.v[s + 8] = lb[(16 + ko + s) * B_STRIDE + bn];
            bf1.v[s]     = lb[(ko + s) * B_STRIDE + bn + 16];
            bf1.v[s + 8] = lb[(16 + ko + s) * B_STRIDE + bn + 16];
        }

        acc00 = __builtin_amdgcn_wmma_f32_16x16x32_bf16(false, a0.v, false, bf0.v,
                                                        (short)0, acc00, false, false);
        acc01 = __builtin_amdgcn_wmma_f32_16x16x32_bf16(false, a0.v, false, bf1.v,
                                                        (short)0, acc01, false, false);
        acc10 = __builtin_amdgcn_wmma_f32_16x16x32_bf16(false, a1.v, false, bf0.v,
                                                        (short)0, acc10, false, false);
        acc11 = __builtin_amdgcn_wmma_f32_16x16x32_bf16(false, a1.v, false, bf1.v,
                                                        (short)0, acc11, false, false);
        __syncthreads();
    }

    // epilogue: D layout — lane n = lane&15, VGPR v -> m = v (+8 for lanes>=16)
    const int nc   = o0 + wn * 32 + (lane & 15);
    const int mofs = (lane < 16) ? 0 : 8;
    #pragma unroll
    for (int v = 0; v < 8; ++v) {
        {
            const int ml = wm * 32 + v + mofs;           // sub-tile 0
            if (m0 + ml < c) {
                const float wv = s_wgt[ml];
                float* op = out + (size_t)s_tok[ml] * DD;
                unsafeAtomicAdd(op + nc,      wv * acc00[v]);
                unsafeAtomicAdd(op + nc + 16, wv * acc01[v]);
            }
        }
        {
            const int ml = wm * 32 + 16 + v + mofs;      // sub-tile 1
            if (m0 + ml < c) {
                const float wv = s_wgt[ml];
                float* op = out + (size_t)s_tok[ml] * DD;
                unsafeAtomicAdd(op + nc,      wv * acc10[v]);
                unsafeAtomicAdd(op + nc + 16, wv * acc11[v]);
            }
        }
    }
}

// ---------------------------------------------------------------------------
// Host launcher
// ---------------------------------------------------------------------------
extern "C" void kernel_launch(void* const* d_in, const int* in_sizes, int n_in,
                              void* d_out, int out_size, void* d_ws, size_t ws_size,
                              hipStream_t stream)
{
    const float* x  = (const float*)d_in[0];   // [N, D]
    const float* W  = (const float*)d_in[1];   // [E, D, D]
    const float* b  = (const float*)d_in[2];   // [E, D]
    const float* Wr = (const float*)d_in[3];   // [E, D]
    const float* br = (const float*)d_in[4];   // [E]
    float* out = (float*)d_out;                // [N, D]

    // workspace layout
    char* ws = (char*)d_ws;
    int*    cnt = (int*)ws;                                      // 8 ints
    int*    tok = (int*)(ws + 128);                              // [E][NTOK]
    float*  wgt = (float*)(ws + 128 + (size_t)NE * NTOK * 4);    // [E][NTOK]
    __bf16* xb  = (__bf16*)(ws + (1u << 20));                    // [N][D] bf16, 32MB
    __bf16* Wt  = (__bf16*)(ws + (1u << 20) +
                            (size_t)NTOK * DD * sizeof(__bf16)); // [E][D][D] bf16, 64MB

    zero_cnt<<<1, 32, 0, stream>>>(cnt);

    cvt_x<<<(NTOK * (DD / 8)) / 256, 256, 0, stream>>>(x, xb);

    dim3 tg(DD / 64, DD / 64, NE);
    cvt_w<<<tg, 256, 0, stream>>>(W, Wt);

    router_kernel<<<NTOK, 256, 0, stream>>>(x, Wr, br, b, out, cnt, tok, wgt);

    dim3 grid(NTOK / MT, DD / NT, NE);   // (64, 32, 8); empty tiles early-exit
    moe_gemm<<<grid, 256, 0, stream>>>(xb, Wt, cnt, tok, wgt, out);
}
